// CausalAttnBlock_62783831933251
// MI455X (gfx1250) — matrix operand order repaired and checked by reading, not compiled, not run on hardware
//
#include <hip/hip_runtime.h>

// ---------------- problem constants ----------------
#define EPSV 1e-6f
// B=2, C=256, T=16, H*W=1024 -> 32768 tokens, per-sample elements = 4,194,304

typedef __attribute__((ext_vector_type(16))) __bf16 v16bf;
typedef __attribute__((ext_vector_type(8)))  float  v8f;

// ---------------- workspace layout (bytes) ----------------
static constexpr size_t OFF_PART = 0;                       // 2*128*2 f32 partial sums
static constexpr size_t OFF_MS   = 2048;                    // mean/rstd per sample
static constexpr size_t OFF_W    = 4096;                    // 4 x 256x256 bf16 weights
static constexpr size_t SZ_W     = 4ull * 65536 * 2;        // 512 KB
static constexpr size_t OFF_HN   = OFF_W + SZ_W;            // hn bf16 [32768,256]
static constexpr size_t SZ_TOKC  = 32768ull * 256 * 2;      // 16 MB
static constexpr size_t OFF_Q    = OFF_HN + SZ_TOKC;
static constexpr size_t OFF_K    = OFF_Q + SZ_TOKC;
static constexpr size_t OFF_V    = OFF_K + SZ_TOKC;
static constexpr size_t OFF_O    = OFF_HN;                  // alias: hn dead after proj

// ---------------- helpers ----------------
__device__ __forceinline__ unsigned short f2bf(float f) {
  union { float f; unsigned u; } v; v.f = f;
  unsigned u = v.u;
  u += 0x7FFFu + ((u >> 16) & 1u);   // round-to-nearest-even
  return (unsigned short)(u >> 16);
}

__device__ __forceinline__ v8f wmma_bf16(v16bf a, v16bf b, v8f c) {
  return __builtin_amdgcn_wmma_f32_16x16x32_bf16(false, a, false, b, (short)0, c,
                                                 false, false);
}

// A tile 16x32 bf16, row-major source [m, k], ld in elements.
// Lane layout (contiguous k, matched by B loaders): lanes 0-15: M=lane, K=0..15; lanes 16-31: M=lane-16, K=16..31
__device__ __forceinline__ v16bf load_A16(const unsigned short* base, int ld,
                                          int m0, int k0, int lane) {
  const unsigned short* row =
      base + (size_t)(m0 + (lane & 15)) * ld + k0 + ((lane & 16) ? 16 : 0);
  union { v16bf v; unsigned short s[16]; } u;
#pragma unroll
  for (int e = 0; e < 8; ++e) u.s[e] = row[e];
#pragma unroll
  for (int e = 0; e < 8; ++e) u.s[8 + e] = row[8 + e];
  return u.v;
}

// B tile 32x16 where B[k][n] = src[n][k] (transposed-from-row-major source).
// Lane layout: lanes 0-15: N=lane, K=0..15; lanes 16-31: N=lane-16, K=16..31.
__device__ __forceinline__ v16bf load_BT16(const unsigned short* base, int ld,
                                           int n0, int k0, int lane) {
  const unsigned short* src =
      base + (size_t)(n0 + (lane & 15)) * ld + k0 + ((lane & 16) ? 16 : 0);
  union { v16bf v; unsigned short s[16]; } u;
#pragma unroll
  for (int e = 0; e < 16; ++e) u.s[e] = src[e];
  return u.v;
}

// B tile 32x16 where B[k][n] = src[k][n] (direct row-major source).
__device__ __forceinline__ v16bf load_Brow16(const unsigned short* base, int ld,
                                             int k0, int n0, int lane) {
  const unsigned short* src =
      base + (size_t)(k0 + ((lane & 16) ? 16 : 0)) * ld + n0 + (lane & 15);
  union { v16bf v; unsigned short s[16]; } u;
#pragma unroll
  for (int e = 0; e < 16; ++e) u.s[e] = src[(size_t)e * ld];
  return u.v;
}

// A tile from f32 attention rows in LDS (16 rows x 1024 keys), convert to bf16.
__device__ __forceinline__ v16bf load_A_lds(const float* sc, int k0, int lane) {
  const float* rowp = sc + (lane & 15) * 1024 + k0 + ((lane & 16) ? 16 : 0);
  union { v16bf v; unsigned short s[16]; } u;
#pragma unroll
  for (int e = 0; e < 8; ++e) u.s[e] = f2bf(rowp[e]);
#pragma unroll
  for (int e = 0; e < 8; ++e) u.s[8 + e] = f2bf(rowp[8 + e]);
  return u.v;
}

// ---------------- kernels ----------------

// Stage 1 of GroupNorm stats: deterministic two-level reduction (no atomics).
__global__ void k_stats_partial(const float* __restrict__ x, float* __restrict__ part) {
  __shared__ float s1[256], s2[256];
  int s = blockIdx.y, blk = blockIdx.x, tid = threadIdx.x;
  const float* xs = x + (size_t)s * 4194304ull + (size_t)blk * 32768ull;
  float a = 0.f, b = 0.f;
#pragma unroll 4
  for (int j = 0; j < 128; ++j) { float v = xs[tid + j * 256]; a += v; b += v * v; }
  s1[tid] = a; s2[tid] = b; __syncthreads();
  for (int off = 128; off > 0; off >>= 1) {
    if (tid < off) { s1[tid] += s1[tid + off]; s2[tid] += s2[tid + off]; }
    __syncthreads();
  }
  if (tid == 0) {
    part[(s * 128 + blk) * 2]     = s1[0];
    part[(s * 128 + blk) * 2 + 1] = s2[0];
  }
}

__global__ void k_stats_final(const float* __restrict__ part, float* __restrict__ ms) {
  __shared__ float s1[128], s2[128];
  int tid = threadIdx.x;
  for (int s = 0; s < 2; ++s) {
    s1[tid] = part[(s * 128 + tid) * 2];
    s2[tid] = part[(s * 128 + tid) * 2 + 1];
    __syncthreads();
    for (int off = 64; off > 0; off >>= 1) {
      if (tid < off) { s1[tid] += s1[tid + off]; s2[tid] += s2[tid + off]; }
      __syncthreads();
    }
    if (tid == 0) {
      const float n = 4194304.0f;
      float mean = s1[0] / n;
      float var  = s2[0] / n - mean * mean;
      ms[s * 2]     = mean;
      ms[s * 2 + 1] = rsqrtf(var + EPSV);
    }
    __syncthreads();
  }
}

// Normalize + layout change [B,C,T,HW] -> bf16 [token, C].
__global__ void k_norm_cast(const float* __restrict__ x, const float* __restrict__ gamma,
                            const float* __restrict__ beta, const float* __restrict__ ms,
                            unsigned short* __restrict__ hn) {
  int i = blockIdx.x * 256 + threadIdx.x;       // 0 .. 8388607
  int hw = i & 1023;
  int t  = (i >> 10) & 15;
  int c  = (i >> 14) & 255;
  int b  = i >> 22;
  float mean = ms[b * 2], rstd = ms[b * 2 + 1];
  float v = (x[i] - mean) * rstd * gamma[c] + beta[c];
  int token = b * 16384 + t * 1024 + hw;
  hn[(size_t)token * 256 + c] = f2bf(v);
}

__global__ void k_cast_w(const float* __restrict__ wq, const float* __restrict__ wk,
                         const float* __restrict__ wv, const float* __restrict__ wo,
                         unsigned short* __restrict__ wbf) {
  int i = blockIdx.x * 256 + threadIdx.x;       // 0 .. 262143
  int which = i >> 16, idx = i & 65535;
  const float* src = which == 0 ? wq : which == 1 ? wk : which == 2 ? wv : wo;
  wbf[i] = f2bf(src[idx]);
}

// Q/K/V projections: out[token,n] = sum_c hn[token,c] * W[n,c] + bias[n]
// (q pre-scaled by C^-0.5 = 1/16). One 16x16 tile per wave, K-loop of 8 WMMAs.
__global__ void __launch_bounds__(128)
k_proj(const unsigned short* __restrict__ hn, const unsigned short* __restrict__ wbf,
       const float* __restrict__ bq, const float* __restrict__ bk,
       const float* __restrict__ bv, unsigned short* __restrict__ qb,
       unsigned short* __restrict__ kb, unsigned short* __restrict__ vb) {
  int lane = threadIdx.x & 31, wid = threadIdx.x >> 5;
  int m0 = blockIdx.x * 16;
  int n0 = (blockIdx.y * 4 + wid) * 16;
  int sel = blockIdx.z;
  const unsigned short* W = wbf + (size_t)sel * 65536;
  const float* bias = sel == 0 ? bq : (sel == 1 ? bk : bv);
  unsigned short* outp = sel == 0 ? qb : (sel == 1 ? kb : vb);
  float scale = sel == 0 ? 0.0625f : 1.0f;
  v8f acc = {};
#pragma unroll
  for (int k0 = 0; k0 < 256; k0 += 32) {
    v16bf a = load_A16(hn, 256, m0, k0, lane);
    v16bf b = load_BT16(W, 256, n0, k0, lane);
    acc = wmma_bf16(a, b, acc);
  }
  int n = n0 + (lane & 15);
  int mh = m0 + ((lane & 16) ? 8 : 0);
  float bn = bias[n];
#pragma unroll
  for (int r = 0; r < 8; ++r)
    outp[(size_t)(mh + r) * 256 + n] = f2bf((acc[r] + bn) * scale);
}

// Attention for one (b,t) item and one 16-query row block.
// Phase 1: S = Q K^T into 16x1024 f32 LDS; Phase 2: softmax; Phase 3: O = attn V.
__global__ void __launch_bounds__(128)
k_attn(const unsigned short* __restrict__ qb, const unsigned short* __restrict__ kb,
       const unsigned short* __restrict__ vb, unsigned short* __restrict__ ob) {
  __shared__ float sc[16 * 1024];               // 64 KB of the 320 KB WGP LDS
  int lane = threadIdx.x & 31, wid = threadIdx.x >> 5;
  int g = blockIdx.y;                           // item in [0,32)
  int qbase = g * 1024 + blockIdx.x * 16;       // global query token base
  int kbase = g * 1024;                         // global key token base

  // Phase 1: scores (64 key tiles split over 4 waves, 8 WMMAs each)
  for (int tile = wid; tile < 64; tile += 4) {
    int n0 = tile * 16;
    v8f acc = {};
#pragma unroll
    for (int k0 = 0; k0 < 256; k0 += 32) {
      v16bf a = load_A16(qb, 256, qbase, k0, lane);
      v16bf b = load_BT16(kb, 256, kbase + n0, k0, lane);
      acc = wmma_bf16(a, b, acc);
    }
    int rbase = (lane & 16) ? 8 : 0;
    int col = n0 + (lane & 15);
#pragma unroll
    for (int r = 0; r < 8; ++r) sc[(rbase + r) * 1024 + col] = acc[r];
  }
  __syncthreads();

  // Phase 2: softmax over keys, each wave owns 4 rows, lanes stride the row
  for (int row = wid; row < 16; row += 4) {
    float* rp = sc + row * 1024;
    float mx = -3.0e38f;
    for (int j = lane; j < 1024; j += 32) mx = fmaxf(mx, rp[j]);
    for (int off = 16; off > 0; off >>= 1) mx = fmaxf(mx, __shfl_xor(mx, off, 32));
    float sum = 0.f;
    for (int j = lane; j < 1024; j += 32) {
      float e = __expf(rp[j] - mx);
      rp[j] = e;
      sum += e;
    }
    for (int off = 16; off > 0; off >>= 1) sum += __shfl_xor(sum, off, 32);
    float inv = 1.0f / sum;
    for (int j = lane; j < 1024; j += 32) rp[j] *= inv;
  }
  __syncthreads();

  // Phase 3: O = attn x V (16 col tiles over 4 waves, 32 WMMAs each)
  for (int ct = wid; ct < 16; ct += 4) {
    int c0 = ct * 16;
    v8f acc = {};
    for (int mk = 0; mk < 1024; mk += 32) {
      v16bf a = load_A_lds(sc, mk, lane);
      v16bf b = load_Brow16(vb, 256, kbase + mk, c0, lane);
      acc = wmma_bf16(a, b, acc);
    }
    int n = c0 + (lane & 15);
    int mh = qbase + ((lane & 16) ? 8 : 0);
#pragma unroll
    for (int r = 0; r < 8; ++r) ob[(size_t)(mh + r) * 256 + n] = f2bf(acc[r]);
  }
}

// Final projection + bias + residual, stored back in x's [B,C,T,HW] layout.
// Token dim is contiguous in the output, so each lane's 8 rows are 8
// consecutive floats.
__global__ void __launch_bounds__(128)
k_final(const unsigned short* __restrict__ ob, const unsigned short* __restrict__ wo,
        const float* __restrict__ bo, const float* __restrict__ x,
        float* __restrict__ out) {
  int lane = threadIdx.x & 31, wid = threadIdx.x >> 5;
  int m0 = blockIdx.x * 16;
  int n0 = (blockIdx.y * 4 + wid) * 16;
  v8f acc = {};
#pragma unroll
  for (int k0 = 0; k0 < 256; k0 += 32) {
    v16bf a = load_A16(ob, 256, m0, k0, lane);
    v16bf b = load_BT16(wo, 256, n0, k0, lane);
    acc = wmma_bf16(a, b, acc);
  }
  int n = n0 + (lane & 15);
  float bn = bo[n];
  int tok0 = m0 + ((lane & 16) ? 8 : 0);
#pragma unroll
  for (int r = 0; r < 8; ++r) {
    int tok = tok0 + r;
    int b = tok >> 14;                 // sample
    int tl = tok & 16383;              // t*1024 + hw
    size_t addr = ((size_t)(b * 256 + n)) * 16384 + tl;
    out[addr] = x[addr] + acc[r] + bn;
  }
}

// ---------------- launcher ----------------
extern "C" void kernel_launch(void* const* d_in, const int* in_sizes, int n_in,
                              void* d_out, int out_size, void* d_ws, size_t ws_size,
                              hipStream_t stream) {
  (void)in_sizes; (void)n_in; (void)out_size; (void)ws_size;
  const float* x     = (const float*)d_in[0];
  const float* gamma = (const float*)d_in[1];
  const float* beta  = (const float*)d_in[2];
  const float* wq    = (const float*)d_in[3];
  const float* bq    = (const float*)d_in[4];
  const float* wk    = (const float*)d_in[5];
  const float* bk    = (const float*)d_in[6];
  const float* wv    = (const float*)d_in[7];
  const float* bv    = (const float*)d_in[8];
  const float* wo    = (const float*)d_in[9];
  const float* bo    = (const float*)d_in[10];
  float* out = (float*)d_out;

  char* ws = (char*)d_ws;
  float* part          = (float*)(ws + OFF_PART);
  float* ms            = (float*)(ws + OFF_MS);
  unsigned short* wbf  = (unsigned short*)(ws + OFF_W);
  unsigned short* hn   = (unsigned short*)(ws + OFF_HN);
  unsigned short* qb   = (unsigned short*)(ws + OFF_Q);
  unsigned short* kb   = (unsigned short*)(ws + OFF_K);
  unsigned short* vb   = (unsigned short*)(ws + OFF_V);
  unsigned short* ob   = (unsigned short*)(ws + OFF_O);   // aliases hn (hn dead by then)

  k_stats_partial<<<dim3(128, 2), 256, 0, stream>>>(x, part);
  k_stats_final<<<1, 128, 0, stream>>>(part, ms);
  k_cast_w<<<1024, 256, 0, stream>>>(wq, wk, wv, wo, wbf);
  k_norm_cast<<<32768, 256, 0, stream>>>(x, gamma, beta, ms, hn);
  k_proj<<<dim3(2048, 4, 3), 128, 0, stream>>>(hn, wbf, bq, bk, bv, qb, kb, vb);
  k_attn<<<dim3(64, 32), 128, 0, stream>>>(qb, kb, vb, ob);
  k_final<<<dim3(2048, 4), 128, 0, stream>>>(ob, wbf + 3 * 65536, bo, x, out);
}
